// SparseMultiHeadAttention_4080218931339
// MI455X (gfx1250) — compile-verified, matrix operand order checked by reading, not run on hardware
//
#include <hip/hip_runtime.h>
#include <hip/hip_bf16.h>

// ---------------------------------------------------------------------------
// Sparse windowed MHA for MI455X (gfx1250), wave32.
//  - v_wmma_f32_16x16x32_f16 for all three GEMM stages (f16 in, f32 accum).
//  - global_load_async_to_lds_b128 (ASYNCcnt DMA) with double-buffered LDS
//    staging so DMA of K-slice i+1 overlaps WMMA of slice i.
//  Phase 0: f32->f16 conversion of x, transposed f16 copies of Wqkv / Wout.
//  Phase 1: deterministic window/slot assignment (no atomics).
//  Phase 2: per-(window,head) fused QKV GEMM + RMS-norm + softmax attention.
//  Phase 3: output projection GEMM 65536x1024x1024.
// Workspace use: ~265 MB (xh, oh, WqkvT, WoutT, winTok).
// ---------------------------------------------------------------------------

typedef __attribute__((ext_vector_type(16))) _Float16 v16h;
typedef __attribute__((ext_vector_type(8)))  float    v8f;

#define NTOK  65536
#define CDIM  1024
#define HEADS 16
#define NWIN  1024

__device__ __forceinline__ v8f wmma16(v16h a, v16h b, v8f c) {
  return __builtin_amdgcn_wmma_f32_16x16x32_f16(false, a, false, b, (short)0, c,
                                                false, false);
}

// Low 32 bits of a generic pointer to LDS == DS byte address (ISA 10.2:
// flat->LDS truncates to addr[31:0]).
__device__ __forceinline__ uint32_t lds_off(const void* p) {
  return (uint32_t)(uintptr_t)p;
}

// Async DMA: 16 bytes per lane, global -> LDS, tracked by ASYNCcnt.
__device__ __forceinline__ void async_b128(const void* g, uint32_t l) {
  asm volatile("global_load_async_to_lds_b128 %0, %1, off"
               :: "v"(l), "v"(g) : "memory");
}
#define WAIT_ASYNC(imm) asm volatile("s_wait_asynccnt " #imm ::: "memory")

// 16x32 f16 fragment load (A layout; B tiles staged transposed so the same
// loader works). ISA 7.12.2: lane l holds row M=l%16; halves 0..7 at
// K=(l/16)*8..+7 and halves 8..15 at K=16+(l/16)*8..+7.
__device__ __forceinline__ v16h ldfrag(const _Float16* base, int ld, int row0,
                                       int k0, int lane) {
  const int m = lane & 15, g = lane >> 4;
  const _Float16* p = base + (size_t)(row0 + m) * ld + k0 + g * 8;
  union { v16h v; float4 f[2]; } u;
  u.f[0] = *reinterpret_cast<const float4*>(p);
  u.f[1] = *reinterpret_cast<const float4*>(p + 16);
  return u.v;
}

// ------------------------------ phase 0 ------------------------------------
__global__ __launch_bounds__(256) void convert_f32_f16(
    const float* __restrict__ in, _Float16* __restrict__ out) {
  size_t i = ((size_t)blockIdx.x * 256 + threadIdx.x) * 4;
  float4 v = *reinterpret_cast<const float4*>(in + i);
  union { _Float16 h[4]; float2 f; } u;
  u.h[0] = (_Float16)v.x; u.h[1] = (_Float16)v.y;
  u.h[2] = (_Float16)v.z; u.h[3] = (_Float16)v.w;
  *reinterpret_cast<float2*>(out + i) = u.f;
}

// out[Cc][R] = (f16) transpose of in[R][Cc]
__global__ __launch_bounds__(256) void transpose_convert(
    const float* __restrict__ in, _Float16* __restrict__ out, int R, int Cc) {
  __shared__ float t[32][33];
  const int c0 = blockIdx.x * 32, r0 = blockIdx.y * 32;
  const int tx = threadIdx.x & 31, ty = threadIdx.x >> 5;
  for (int i = ty; i < 32; i += 8)
    t[i][tx] = in[(size_t)(r0 + i) * Cc + c0 + tx];
  __syncthreads();
  for (int i = ty; i < 32; i += 8)
    out[(size_t)(c0 + i) * R + r0 + tx] = (_Float16)t[tx][i];
}

// ------------------------------ phase 1 ------------------------------------
__global__ __launch_bounds__(256) void winmap(
    const int* __restrict__ coords, int* __restrict__ winTok) {
  const int n = blockIdx.x * 256 + threadIdx.x;
  const int4 c = reinterpret_cast<const int4*>(coords)[n];
  const int wid  = ((c.x * 8 + (c.y >> 2)) * 8 + (c.z >> 2)) * 8 + (c.w >> 2);
  const int slot = ((c.y & 3) << 4) | ((c.z & 3) << 2) | (c.w & 3);
  winTok[wid * 64 + slot] = n;
}

// ------------------------------ phase 2 ------------------------------------
// block = 256 threads (8 waves), one block per (window, head).
struct Stage { _Float16 A[64 * 32]; _Float16 B[192 * 32]; };  // 16 KB

__global__ __launch_bounds__(256) void fused_qkv_attn(
    const _Float16* __restrict__ xh, const _Float16* __restrict__ WqkvT,
    const float* __restrict__ bqkv, const float* __restrict__ gq,
    const float* __restrict__ gk, const int* __restrict__ winTok,
    _Float16* __restrict__ oh) {
  __shared__ __align__(16) union UShared {
    Stage st[2];                                               // 32 KB staging
    struct { float scores[64 * 64]; _Float16 attn[64 * 64]; } sm;  // 24 KB
  } uS;
  __shared__ __align__(16) _Float16 sQ[64 * 64];
  __shared__ __align__(16) _Float16 sK[64 * 64];
  __shared__ __align__(16) _Float16 sVT[64 * 64];  // v transposed [d][n]
  __shared__ float sKsq[64][2];
  __shared__ int   sTok[64];
  __shared__ float sGq[64], sGk[64];

  const int tid  = threadIdx.x;
  const int lane = tid & 31;
  const int wv   = tid >> 5;
  const int g    = lane >> 4;
  const int nn   = lane & 15;
  const int w    = blockIdx.x >> 4;
  const int h    = blockIdx.x & 15;
  const int mrow = wv >> 1;        // M tile 0..3
  const int nbas = (wv & 1) * 6;   // 6 of 12 N tiles per wave

  if (tid < 64) {
    sTok[tid] = winTok[w * 64 + tid];
    sGq[tid]  = gq[h * 64 + tid];
    sGk[tid]  = gk[h * 64 + tid];
  }
  __syncthreads();

  // -- loop-invariant staging addresses (async DMA descriptors per thread) --
  const int aRow = tid >> 2, aSeg = tid & 3;
  const _Float16* gA = xh + (size_t)sTok[aRow] * CDIM + aSeg * 8;
  uint32_t lA[2] = {lds_off(&uS.st[0].A[aRow * 32 + aSeg * 8]),
                    lds_off(&uS.st[1].A[aRow * 32 + aSeg * 8])};
  const _Float16* gB[3];
  uint32_t lB[2][3];
#pragma unroll
  for (int i = 0; i < 3; ++i) {
    const int vid = tid + i * 256;
    const int n = vid >> 2, seg = vid & 3;
    const int col = (n >> 6) * CDIM + h * 64 + (n & 63);  // q|k|v column
    gB[i] = WqkvT + (size_t)col * CDIM + seg * 8;
    lB[0][i] = lds_off(&uS.st[0].B[n * 32 + seg * 8]);
    lB[1][i] = lds_off(&uS.st[1].B[n * 32 + seg * 8]);
  }

  // ---- QKV GEMM: [64 x 1024] x [1024 x 192], double-buffered async staging -
  v8f acc[6] = {};
  async_b128(gA, lA[0]);
#pragma unroll
  for (int i = 0; i < 3; ++i) async_b128(gB[i], lB[0][i]);

  for (int kk = 0; kk < 1024; kk += 32) {
    const int p = (kk >> 5) & 1;
    if (kk + 32 < 1024) {  // prefetch next K slice into the other buffer
      const int np = p ^ 1;
      async_b128(gA + kk + 32, lA[np]);
#pragma unroll
      for (int i = 0; i < 3; ++i) async_b128(gB[i] + kk + 32, lB[np][i]);
      WAIT_ASYNC(0x4);  // 4 newest in flight; current buffer complete
    } else {
      WAIT_ASYNC(0x0);
    }
    __syncthreads();
    v16h a = ldfrag(uS.st[p].A, 32, mrow * 16, 0, lane);
    v16h bf[6];
#pragma unroll
    for (int j = 0; j < 6; ++j)
      bf[j] = ldfrag(uS.st[p].B, 32, (nbas + j) * 16, 0, lane);
#pragma unroll
    for (int j = 0; j < 6; ++j) acc[j] = wmma16(a, bf[j], acc[j]);
    __syncthreads();
  }

  // ---- + bqkv --------------------------------------------------------------
#pragma unroll
  for (int j = 0; j < 6; ++j) {
    const int nt  = nbas + j;  // 0..11: q tiles 0-3, k 4-7, v 8-11
    const float b = bqkv[(nt >> 2) * CDIM + h * 64 + (nt & 3) * 16 + nn];
#pragma unroll
    for (int r = 0; r < 8; ++r) acc[j][r] += b;
  }

  // ---- RMS-norm sums (per row of 64): shuffle across 16 lanes --------------
  float qinv[8];
  if (nbas == 0) {  // even waves: all q cols + k cols 0..31
#pragma unroll
    for (int r = 0; r < 8; ++r) {
      float qs = 0.f;
#pragma unroll
      for (int j = 0; j < 4; ++j) qs += acc[j][r] * acc[j][r];
      float ks = acc[4][r] * acc[4][r] + acc[5][r] * acc[5][r];
#pragma unroll
      for (int m = 1; m < 16; m <<= 1) {
        qs += __shfl_xor(qs, m, 32);
        ks += __shfl_xor(ks, m, 32);
      }
      qinv[r] = 8.0f / fmaxf(sqrtf(qs), 1e-12f);  // sqrt(D)/||t||
      if (nn == 0) sKsq[mrow * 16 + g * 8 + r][0] = ks;
    }
  } else {  // odd waves: k cols 32..63 + all v
#pragma unroll
    for (int r = 0; r < 8; ++r) {
      float ks = acc[0][r] * acc[0][r] + acc[1][r] * acc[1][r];
#pragma unroll
      for (int m = 1; m < 16; m <<= 1) ks += __shfl_xor(ks, m, 32);
      if (nn == 0) sKsq[mrow * 16 + g * 8 + r][1] = ks;
    }
  }
  __syncthreads();

  // ---- scale + pack f16 tiles ----------------------------------------------
  if (nbas == 0) {
#pragma unroll
    for (int r = 0; r < 8; ++r) {
      const int row = mrow * 16 + g * 8 + r;
      const float kinv = 8.0f / fmaxf(sqrtf(sKsq[row][0] + sKsq[row][1]), 1e-12f);
#pragma unroll
      for (int j = 0; j < 4; ++j) {
        const int col = j * 16 + nn;
        sQ[row * 64 + col] = (_Float16)(acc[j][r] * qinv[r] * sGq[col]);
      }
#pragma unroll
      for (int j = 4; j < 6; ++j) {
        const int col = (j - 4) * 16 + nn;
        sK[row * 64 + col] = (_Float16)(acc[j][r] * kinv * sGk[col]);
      }
    }
  } else {
#pragma unroll
    for (int r = 0; r < 8; ++r) {
      const int row = mrow * 16 + g * 8 + r;
      const float kinv = 8.0f / fmaxf(sqrtf(sKsq[row][0] + sKsq[row][1]), 1e-12f);
#pragma unroll
      for (int j = 0; j < 2; ++j) {
        const int col = 32 + j * 16 + nn;
        sK[row * 64 + col] = (_Float16)(acc[j][r] * kinv * sGk[col]);
      }
#pragma unroll
      for (int j = 2; j < 6; ++j) {
        const int d = (j - 2) * 16 + nn;
        sVT[d * 64 + row] = (_Float16)acc[j][r];
      }
    }
  }
  __syncthreads();

  // ---- scores = q k^T / sqrt(D): 4x4 tiles, 2 per wave ---------------------
#pragma unroll
  for (int i = 0; i < 2; ++i) {
    const int t = wv * 2 + i, sm = t >> 2, sn = t & 3;
    v8f sc = {};
#pragma unroll
    for (int kk = 0; kk < 64; kk += 32) {
      v16h a = ldfrag(sQ, 64, sm * 16, kk, lane);
      v16h b = ldfrag(sK, 64, sn * 16, kk, lane);
      sc = wmma16(a, b, sc);
    }
#pragma unroll
    for (int r = 0; r < 8; ++r)
      uS.sm.scores[(sm * 16 + g * 8 + r) * 64 + sn * 16 + nn] = sc[r] * 0.125f;
  }
  __syncthreads();

  // ---- softmax rows -> f16 attn --------------------------------------------
  if (tid < 64) {
    float mx = -1e30f;
    for (int c = 0; c < 64; ++c) mx = fmaxf(mx, uS.sm.scores[tid * 64 + c]);
    float sum = 0.f;
    for (int c = 0; c < 64; ++c) {
      const float e = __expf(uS.sm.scores[tid * 64 + c] - mx);
      uS.sm.scores[tid * 64 + c] = e;
      sum += e;
    }
    const float inv = 1.0f / sum;
    for (int c = 0; c < 64; ++c)
      uS.sm.attn[tid * 64 + c] = (_Float16)(uS.sm.scores[tid * 64 + c] * inv);
  }
  __syncthreads();

  // ---- o = attn @ v, scatter rows to original token order ------------------
#pragma unroll
  for (int i = 0; i < 2; ++i) {
    const int t = wv * 2 + i, om = t >> 2, on = t & 3;
    v8f oa = {};
#pragma unroll
    for (int kk = 0; kk < 64; kk += 32) {
      v16h a = ldfrag(uS.sm.attn, 64, om * 16, kk, lane);
      v16h b = ldfrag(sVT, 64, on * 16, kk, lane);
      oa = wmma16(a, b, oa);
    }
#pragma unroll
    for (int r = 0; r < 8; ++r) {
      const int row = om * 16 + g * 8 + r;
      const int col = on * 16 + nn;
      oh[(size_t)sTok[row] * CDIM + h * 64 + col] = (_Float16)oa[r];
    }
  }
}

// ------------------------------ phase 3 ------------------------------------
// out[N,1024] = oh[N,1024](f16) @ Wout(f16) + bout ; block tile 128x64.
struct StageO { _Float16 A[128 * 32]; _Float16 B[64 * 32]; };  // 12 KB

__global__ __launch_bounds__(256) void gemm_out(
    const _Float16* __restrict__ oh, const _Float16* __restrict__ WoutT,
    const float* __restrict__ bout, float* __restrict__ out) {
  __shared__ __align__(16) StageO st[2];  // 24 KB double buffer
  const int tid = threadIdx.x, lane = tid & 31, wv = tid >> 5;
  const int g = lane >> 4, nn = lane & 15;
  const int col0 = (blockIdx.x & 15) * 64;
  const int row0 = (blockIdx.x >> 4) * 128;

  // loop-invariant staging addresses (2 A vectors + 1 B vector per thread)
  const _Float16* gA[2];
  uint32_t lA[2][2];
#pragma unroll
  for (int i = 0; i < 2; ++i) {
    const int vid = tid * 2 + i;
    const int row = vid >> 2, seg = vid & 3;
    gA[i] = oh + (size_t)(row0 + row) * CDIM + seg * 8;
    lA[0][i] = lds_off(&st[0].A[row * 32 + seg * 8]);
    lA[1][i] = lds_off(&st[1].A[row * 32 + seg * 8]);
  }
  const int bn = tid >> 2, bseg = tid & 3;
  const _Float16* gB = WoutT + (size_t)(col0 + bn) * CDIM + bseg * 8;
  uint32_t lB[2] = {lds_off(&st[0].B[bn * 32 + bseg * 8]),
                    lds_off(&st[1].B[bn * 32 + bseg * 8])};

  v8f acc[4] = {};
  async_b128(gA[0], lA[0][0]);
  async_b128(gA[1], lA[0][1]);
  async_b128(gB, lB[0]);

  for (int kk = 0; kk < 1024; kk += 32) {
    const int p = (kk >> 5) & 1;
    if (kk + 32 < 1024) {
      const int np = p ^ 1;
      async_b128(gA[0] + kk + 32, lA[np][0]);
      async_b128(gA[1] + kk + 32, lA[np][1]);
      async_b128(gB + kk + 32, lB[np]);
      WAIT_ASYNC(0x3);
    } else {
      WAIT_ASYNC(0x0);
    }
    __syncthreads();
    v16h a = ldfrag(st[p].A, 32, wv * 16, 0, lane);
    v16h bf[4];
#pragma unroll
    for (int j = 0; j < 4; ++j) bf[j] = ldfrag(st[p].B, 32, j * 16, 0, lane);
#pragma unroll
    for (int j = 0; j < 4; ++j) acc[j] = wmma16(a, bf[j], acc[j]);
    __syncthreads();
  }
#pragma unroll
  for (int j = 0; j < 4; ++j)
#pragma unroll
    for (int r = 0; r < 8; ++r) {
      const int row = row0 + wv * 16 + g * 8 + r;
      const int col = col0 + j * 16 + nn;
      out[(size_t)row * CDIM + col] = acc[j][r] + bout[col];
    }
}

// ---------------------------------------------------------------------------
extern "C" void kernel_launch(void* const* d_in, const int* in_sizes, int n_in,
                              void* d_out, int out_size, void* d_ws,
                              size_t ws_size, hipStream_t stream) {
  const float* x      = (const float*)d_in[0];
  const int*   coords = (const int*)d_in[1];
  const float* Wqkv   = (const float*)d_in[2];
  const float* bqkv   = (const float*)d_in[3];
  const float* gq     = (const float*)d_in[4];
  const float* gk     = (const float*)d_in[5];
  const float* Wout   = (const float*)d_in[6];
  const float* bout   = (const float*)d_in[7];
  float*       out    = (float*)d_out;

  char* p = (char*)d_ws;  // ~265 MB total
  _Float16* xh    = (_Float16*)p; p += (size_t)NTOK * CDIM * 2;
  _Float16* oh    = (_Float16*)p; p += (size_t)NTOK * CDIM * 2;
  _Float16* WqkvT = (_Float16*)p; p += (size_t)3 * CDIM * CDIM * 2;
  _Float16* WoutT = (_Float16*)p; p += (size_t)CDIM * CDIM * 2;
  int*      winTok = (int*)p;

  convert_f32_f16<<<(size_t)NTOK * CDIM / 1024, 256, 0, stream>>>(x, xh);
  transpose_convert<<<dim3(3 * CDIM / 32, CDIM / 32), 256, 0, stream>>>(
      Wqkv, WqkvT, CDIM, 3 * CDIM);
  transpose_convert<<<dim3(CDIM / 32, CDIM / 32), 256, 0, stream>>>(
      Wout, WoutT, CDIM, CDIM);
  winmap<<<NTOK / 256, 256, 0, stream>>>(coords, winTok);
  fused_qkv_attn<<<NWIN * HEADS, 256, 0, stream>>>(xh, WqkvT, bqkv, gq, gk,
                                                   winTok, oh);
  gemm_out<<<(NTOK / 128) * (CDIM / 64), 256, 0, stream>>>(oh, WoutT, bout,
                                                           out);
}
